// MultiHeadAttention_45380624449902
// MI455X (gfx1250) — matrix operand order repaired and checked by reading, not compiled, not run on hardware
//
#include <hip/hip_runtime.h>
#include <hip/hip_bf16.h>

// ---------------- problem constants ----------------
#define BB 2
#define SS 2048
#define DD 1024
#define HH 16
#define HD 64
#define MM (BB*SS)          // 4096 rows for the projection GEMMs

// ---------------- vector types ----------------
typedef _Float16 half8  __attribute__((ext_vector_type(8)));
typedef _Float16 half16 __attribute__((ext_vector_type(16)));
typedef float    floatx8 __attribute__((ext_vector_type(8)));

union H16 { half16 v; half8 h[2]; };

__device__ __forceinline__ floatx8 wmma_f16(half16 a, half16 b, floatx8 c) {
    return __builtin_amdgcn_wmma_f32_16x16x32_f16(false, a, false, b, (short)0, c, false, false);
}

// flat shared pointer -> 32-bit LDS byte address (LDS aperture keeps offset in addr[31:0])
__device__ __forceinline__ unsigned lds_addr32(const void* p) {
    return (unsigned)(size_t)p;
}

// async memory -> LDS copy, 16 bytes per lane (ASYNCcnt-tracked)
__device__ __forceinline__ void async_copy_b128(void* lds_dst, const void* gsrc) {
    unsigned l = lds_addr32(lds_dst);
    asm volatile("global_load_async_to_lds_b128 %0, %1, off"
                 :: "v"(l), "v"(gsrc) : "memory");
}
__device__ __forceinline__ void wait_async0() {
    asm volatile("s_wait_asynccnt 0x0" ::: "memory");
}

// ---------------- fp32 -> fp16 convert ----------------
__global__ void cvt_f32_to_f16(const float* __restrict__ src, _Float16* __restrict__ dst, int n) {
    int i = blockIdx.x * blockDim.x + threadIdx.x;
    if (i < n) dst[i] = (_Float16)src[i];
}

// ---------------- WMMA GEMM: C(MxN) = A(MxK,f16) * B(KxN,f16) ----------------
// 128x64 block tile, 256 threads (8 waves, 4x2), each wave 32x32.
// Double-buffered LDS; A tile filled by async global->LDS DMA.
// MODE 1: f32 output + bias (final projection)
// MODE 2: f16 output with fused RoPE (Q/K projections)
// MODE 3: f16 output stored transposed (b,d,s) (V projection)
#define LDSB_STRIDE 40
template <int MODE>
__global__ __launch_bounds__(256)
void gemm_wmma(const _Float16* __restrict__ A, const _Float16* __restrict__ Bw,
               _Float16* __restrict__ Ch, float* __restrict__ Cf,
               const float* __restrict__ bias,
               const float* __restrict__ pcos, const float* __restrict__ psin,
               int M, int N, int K) {
    __shared__ __attribute__((aligned(16))) _Float16 ldsA[2][128 * 32];
    __shared__ __attribute__((aligned(16))) _Float16 ldsBt[2][64 * LDSB_STRIDE];
    const int tid  = threadIdx.x;
    const int lane = tid & 31;
    const int wave = tid >> 5;
    const int wm = wave >> 1, wn = wave & 1;       // 4 x 2 wave grid
    const int m0 = blockIdx.y * 128;
    const int n0 = blockIdx.x * 64;
    const int lrow = lane & 15;
    const int hsel = lane >> 4;

    // tile fill helpers (inlined lambdas)
    auto fillA = [&](int buf, int kc) {
        #pragma unroll
        for (int i = 0; i < 2; ++i) {
            int e8 = tid + i * 256;                 // 512 x 16B chunks
            int r   = e8 >> 2;                      // 0..127
            int col = (e8 & 3) * 8;
            async_copy_b128(&ldsA[buf][r * 32 + col],
                            &A[(size_t)(m0 + r) * K + kc + col]);
        }
    };
    auto fillB = [&](int buf, int kc) {
        #pragma unroll
        for (int i = 0; i < 8; ++i) {
            int e = tid + i * 256;
            int r   = e >> 6;                       // 0..31 (K)
            int col = e & 63;                       // N
            ldsBt[buf][col * LDSB_STRIDE + r] = Bw[(size_t)(kc + r) * N + n0 + col];
        }
    };

    floatx8 c[2][2] = {};

    // prologue: fill buffer 0
    fillA(0, 0);
    fillB(0, 0);
    wait_async0();
    __syncthreads();

    int cur = 0;
    for (int kc = 0; kc < K; kc += 32) {
        const int nxt = cur ^ 1;
        // start DMA of next tiles while computing on current
        if (kc + 32 < K) {
            fillA(nxt, kc + 32);
            fillB(nxt, kc + 32);
        }
        // prefetch B two chunks ahead (global_prefetch_b8)
        if (kc + 64 < K)
            __builtin_prefetch(&Bw[(size_t)(kc + 64 + (tid >> 3)) * N + n0 + (tid & 7) * 8], 0, 0);

        half16 afrag[2], bfrag[2];
        #pragma unroll
        for (int i = 0; i < 2; ++i) {
            int m = wm * 32 + i * 16 + lrow;
            H16 t;
            t.h[0] = *(const half8*)&ldsA[cur][m * 32 + hsel * 8];
            t.h[1] = *(const half8*)&ldsA[cur][m * 32 + 16 + hsel * 8];
            afrag[i] = t.v;
        }
        #pragma unroll
        for (int j = 0; j < 2; ++j) {
            int n = wn * 32 + j * 16 + lrow;
            H16 t;
            t.h[0] = *(const half8*)&ldsBt[cur][n * LDSB_STRIDE + hsel * 8];
            t.h[1] = *(const half8*)&ldsBt[cur][n * LDSB_STRIDE + 16 + hsel * 8];
            bfrag[j] = t.v;
        }
        #pragma unroll
        for (int i = 0; i < 2; ++i)
            #pragma unroll
            for (int j = 0; j < 2; ++j)
                c[i][j] = wmma_f16(afrag[i], bfrag[j], c[i][j]);

        wait_async0();          // next-buffer DMA complete (this wave's share)
        __syncthreads();        // all waves' shares visible
        cur = nxt;
    }

    #pragma unroll
    for (int i = 0; i < 2; ++i)
        #pragma unroll
        for (int j = 0; j < 2; ++j) {
            int colg = n0 + wn * 32 + j * 16 + lrow;
            #pragma unroll
            for (int r = 0; r < 8; ++r) {
                int rowg = m0 + wm * 32 + i * 16 + r + 8 * hsel;
                if (MODE == 1) {
                    Cf[(size_t)rowg * N + colg] = c[i][j][r] + bias[colg];
                } else if (MODE == 2) {
                    // fused RoPE: pair partner lives in the adjacent lane
                    float v  = c[i][j][r];
                    float pv = __shfl_xor(v, 1, 32);
                    int s  = rowg & (SS - 1);
                    int fi = (colg & 63) >> 1;
                    float co = pcos[s * 32 + fi];
                    float sn = psin[s * 32 + fi];
                    float res = (colg & 1) ? (pv * sn + v * co)   // odd: te*s + to*c
                                           : (v * co - pv * sn);  // even: te*c - to*s
                    Ch[(size_t)rowg * N + colg] = (_Float16)res;
                } else { // MODE 3: store V transposed -> vt[b][d][s]
                    int s = rowg & (SS - 1);
                    int b = rowg >> 11;
                    Ch[((size_t)b * DD + colg) * SS + s] = (_Float16)c[i][j][r];
                }
            }
        }
}

// ---------------- Flash attention: 1 wave per 16-query tile per (b,h) ----------------
// 64-key blocks: 8 score WMMAs + 8 PV WMMAs per block.
__global__ __launch_bounds__(32)
void attn_kernel(const _Float16* __restrict__ qh, const _Float16* __restrict__ kh,
                 const _Float16* __restrict__ vt, _Float16* __restrict__ ctxh) {
    __shared__ __attribute__((aligned(16))) _Float16 ldsP[16 * 64];
    const int lane = threadIdx.x & 31;
    const int lrow = lane & 15, hsel = lane >> 4;
    const int qt = blockIdx.x;
    const int h  = blockIdx.y;
    const int b  = blockIdx.z;
    const int qbase = qt * 16;

    // Q A-fragments: rows = query positions, K = head dim (two 32-chunks)
    half16 aQ[2];
    {
        size_t qrow = ((size_t)b * SS + qbase + lrow) * DD + h * HD;
        #pragma unroll
        for (int ch = 0; ch < 2; ++ch) {
            H16 t;
            t.h[0] = *(const half8*)&qh[qrow + ch * 32 + hsel * 8];
            t.h[1] = *(const half8*)&qh[qrow + ch * 32 + 16 + hsel * 8];
            aQ[ch] = t.v;
        }
    }

    floatx8 o[4] = {};
    float mrow[8], lsum[8];
    #pragma unroll
    for (int r = 0; r < 8; ++r) { mrow[r] = -1e30f; lsum[r] = 0.f; }

    const int kbmax = (qbase + 15) >> 6;      // inclusive, blocks of 64 keys
    const float sscale = 0.125f;              // 1/sqrt(64)

    for (int kb = 0; kb <= kbmax; ++kb) {
        const int kpos0 = kb * 64;

        // scores: four 16x16 tiles, each = sum of two K=32 WMMAs
        floatx8 sc[4];
        #pragma unroll
        for (int t16 = 0; t16 < 4; ++t16) {
            int key = kpos0 + t16 * 16 + lrow;          // lane's key column
            size_t krow = ((size_t)b * SS + key) * DD + h * HD;
            floatx8 acc = {};
            #pragma unroll
            for (int ch = 0; ch < 2; ++ch) {
                H16 t;
                t.h[0] = *(const half8*)&kh[krow + ch * 32 + hsel * 8];
                t.h[1] = *(const half8*)&kh[krow + ch * 32 + 16 + hsel * 8];
                acc = wmma_f16(aQ[ch], t.v, acc);
            }
            sc[t16] = acc;
        }

        const bool needmask = (kpos0 + 63 > qbase);  // wave-uniform

        // online softmax update (per-lane rows r + 8*hsel)
        float pscale[8];
        #pragma unroll
        for (int r = 0; r < 8; ++r) {
            int q = qbase + r + 8 * hsel;
            float v[4];
            #pragma unroll
            for (int t = 0; t < 4; ++t) {
                v[t] = sc[t][r] * sscale;
                if (needmask && (kpos0 + t * 16 + lrow > q)) v[t] = -1e30f;
            }
            float mx = fmaxf(fmaxf(v[0], v[1]), fmaxf(v[2], v[3]));
            #pragma unroll
            for (int off = 1; off < 16; off <<= 1)
                mx = fmaxf(mx, __shfl_xor(mx, off, 32));
            float mnew = fmaxf(mrow[r], mx);
            float sold = __expf(mrow[r] - mnew);
            mrow[r] = mnew;
            float p[4], rs = 0.f;
            #pragma unroll
            for (int t = 0; t < 4; ++t) { p[t] = __expf(v[t] - mnew); rs += p[t]; }
            #pragma unroll
            for (int off = 1; off < 16; off <<= 1)
                rs += __shfl_xor(rs, off, 32);
            lsum[r] = lsum[r] * sold + rs;
            pscale[r] = sold;
            #pragma unroll
            for (int t = 0; t < 4; ++t)
                ldsP[(r + 8 * hsel) * 64 + t * 16 + lrow] = (_Float16)p[t];
        }
        __syncthreads();

        // P as two A-fragments (16 rows x 64 keys)
        half16 aP[2];
        #pragma unroll
        for (int ch = 0; ch < 2; ++ch) {
            H16 tp;
            tp.h[0] = *(const half8*)&ldsP[lrow * 64 + ch * 32 + hsel * 8];
            tp.h[1] = *(const half8*)&ldsP[lrow * 64 + ch * 32 + 16 + hsel * 8];
            aP[ch] = tp.v;
        }

        // O = O*scale + P @ V  (4 head-dim n-tiles x 2 key-chunks)
        #pragma unroll
        for (int nt = 0; nt < 4; ++nt) {
            #pragma unroll
            for (int r = 0; r < 8; ++r) o[nt][r] *= pscale[r];
            #pragma unroll
            for (int ch = 0; ch < 2; ++ch) {
                const _Float16* vb =
                    &vt[((size_t)b * DD + h * HD + nt * 16 + lrow) * SS +
                        kpos0 + ch * 32 + hsel * 8];
                H16 tv;
                tv.h[0] = *(const half8*)vb;
                tv.h[1] = *(const half8*)(vb + 16);
                o[nt] = wmma_f16(aP[ch], tv.v, o[nt]);
            }
        }
        __syncthreads();
    }

    // normalize and store context (f16, row-major (b,s,d))
    float linv[8];
    #pragma unroll
    for (int r = 0; r < 8; ++r) linv[r] = 1.0f / lsum[r];
    #pragma unroll
    for (int nt = 0; nt < 4; ++nt) {
        int col = h * HD + nt * 16 + lrow;
        #pragma unroll
        for (int r = 0; r < 8; ++r) {
            int q = qbase + r + 8 * hsel;
            ctxh[((size_t)b * SS + q) * DD + col] = (_Float16)(o[nt][r] * linv[r]);
        }
    }
}

// ---------------- host launcher ----------------
extern "C" void kernel_launch(void* const* d_in, const int* in_sizes, int n_in,
                              void* d_out, int out_size, void* d_ws, size_t ws_size,
                              hipStream_t stream) {
    const float* x    = (const float*)d_in[0];
    const float* pcos = (const float*)d_in[1];
    const float* psin = (const float*)d_in[2];
    const float* Wq   = (const float*)d_in[3];
    const float* Wk   = (const float*)d_in[4];
    const float* Wv   = (const float*)d_in[5];
    const float* Wo   = (const float*)d_in[6];
    const float* bo   = (const float*)d_in[7];
    float* out = (float*)d_out;

    const size_t NH = (size_t)1024 * 1024;   // 1M halves
    _Float16* ws   = (_Float16*)d_ws;
    _Float16* xh   = ws;                // 4M halves
    _Float16* wqh  = ws + 4 * NH;
    _Float16* wkh  = ws + 5 * NH;
    _Float16* wvh  = ws + 6 * NH;
    _Float16* woh  = ws + 7 * NH;
    _Float16* qh   = ws + 8 * NH;       // 4M (RoPE'd)
    _Float16* kh   = ws + 12 * NH;      // 4M (RoPE'd)
    _Float16* vt   = ws + 16 * NH;      // 4M (transposed (b,d,s))
    _Float16* ctxh = ws + 20 * NH;      // 4M  (total 48 MB)

    const int nx = BB * SS * DD;        // 4M
    const int nw = DD * DD;             // 1M

    cvt_f32_to_f16<<<(nx + 255) / 256, 256, 0, stream>>>(x,  xh,  nx);
    cvt_f32_to_f16<<<(nw + 255) / 256, 256, 0, stream>>>(Wq, wqh, nw);
    cvt_f32_to_f16<<<(nw + 255) / 256, 256, 0, stream>>>(Wk, wkh, nw);
    cvt_f32_to_f16<<<(nw + 255) / 256, 256, 0, stream>>>(Wv, wvh, nw);
    cvt_f32_to_f16<<<(nw + 255) / 256, 256, 0, stream>>>(Wo, woh, nw);

    dim3 ggrid(DD / 64, MM / 128);      // (16, 32)
    // Q, K with fused RoPE; V stored transposed
    gemm_wmma<2><<<ggrid, 256, 0, stream>>>(xh, wqh, qh, nullptr, nullptr, pcos, psin, MM, DD, DD);
    gemm_wmma<2><<<ggrid, 256, 0, stream>>>(xh, wkh, kh, nullptr, nullptr, pcos, psin, MM, DD, DD);
    gemm_wmma<3><<<ggrid, 256, 0, stream>>>(xh, wvh, vt, nullptr, nullptr, nullptr, nullptr, MM, DD, DD);

    dim3 agrid(SS / 16, HH, BB);        // (128, 16, 2)
    attn_kernel<<<agrid, 32, 0, stream>>>(qh, kh, vt, ctxh);

    gemm_wmma<1><<<ggrid, 256, 0, stream>>>(ctxh, woh, nullptr, out, bo, nullptr, nullptr, MM, DD, DD);
}
